// Model_15908558865579
// MI455X (gfx1250) — compile-verified, matrix operand order checked by reading, not run on hardware
//
#include <hip/hip_runtime.h>

#define NHEAD 16
#define SEQ   2048
#define DHEAD 128
#define SCALE 0.022097086912079608f   // 1/sqrt(16*128)
#define LDSW  (SEQ + 24)              // bf16 row pitch: 2072 hw = 4144B, 16B-aligned,
                                      // dword-stride mod 64 = 12 -> conflict-free 16B loads

typedef __attribute__((ext_vector_type(16))) __bf16 v16bf;
typedef __attribute__((ext_vector_type(8)))  float  v8f;

// 16x32 bf16 fragment with K contiguous in memory (A from x1, B from x2 since
// B(k,n)=x2[n][k]).  ISA 7.12.2: lanes 0-15 kb=0, lanes 16-31 kb=8;
// elems 0..7 -> K=kb+j, elems 8..15 -> K=kb+16+(j-8).
__device__ __forceinline__ v16bf gather_rowmajor(const float* __restrict__ base,
                                                 int stride, int fixedIdx,
                                                 int k0, int lane) {
  const int kb = (lane & 16) ? 8 : 0;
  const float* p = base + (size_t)fixedIdx * stride + k0 + kb;
  v16bf f;
#pragma unroll
  for (int j = 0; j < 8; ++j) f[j] = (__bf16)p[j];
#pragma unroll
  for (int j = 0; j < 8; ++j) f[8 + j] = (__bf16)p[16 + j];
  return f;
}

// 32x16 B fragment with K as the row index (x3 as second-GEMM B operand):
// per element-slot j the 16 lanes read 64B contiguous -> coalesced.
__device__ __forceinline__ v16bf gather_colmajor(const float* __restrict__ base,
                                                 int ldm, int n, int k0, int lane) {
  const int kb = (lane & 16) ? 8 : 0;
  const float* p = base + (size_t)(k0 + kb) * ldm + n;
  v16bf f;
#pragma unroll
  for (int j = 0; j < 8; ++j) f[j] = (__bf16)p[(size_t)j * ldm];
#pragma unroll
  for (int j = 0; j < 8; ++j) f[8 + j] = (__bf16)p[(size_t)(16 + j) * ldm];
  return f;
}

__global__ __launch_bounds__(256)
void fused_attn_bias_gemm(const float* __restrict__ x1,
                          const float* __restrict__ x2,
                          const float* __restrict__ x3,
                          float* __restrict__ out) {
  const int head    = blockIdx.y;
  const int rowBase = blockIdx.x * 16;          // 16 query rows per workgroup
  const int tid     = threadIdx.x;
  const int wave    = tid >> 5;
  const int lane    = tid & 31;
  const int colBase = wave * 256;               // 8 waves x 256 cols = 2048

  const float* x1h = x1 + (size_t)head * SEQ * DHEAD;
  const float* x2h = x2 + (size_t)head * SEQ * DHEAD;
  const float* x3h = x3 + (size_t)head * SEQ * SEQ;
  float*       oh  = out + (size_t)head * SEQ * SEQ;  // (h*2048+s)*2048 + t

  __shared__ __bf16 sc[16][LDSW];               // scores / attn row-block (bf16, ~66KB)
  __shared__ float  red[16][16];                // softmax reductions

  const int nc = lane & 15;                     // N (or M) sub-index
  const int rb = (lane & 16) ? 8 : 0;           // C/D row split
  const int kb = (lane & 16) ? 8 : 0;           // A/B K split

  // ---------------- Phase A: scores = x1 * x2^T * scale + x3 ----------------
  v8f acc[16];
  {
    v8f z = {};
#pragma unroll
    for (int t = 0; t < 16; ++t) acc[t] = z;
  }

#pragma unroll
  for (int k0 = 0; k0 < DHEAD; k0 += 32) {
    v16bf a = gather_rowmajor(x1h, DHEAD, rowBase + nc, k0, lane);
    v16bf bb[2];
    bb[0] = gather_rowmajor(x2h, DHEAD, colBase + nc, k0, lane);
#pragma unroll
    for (int t = 0; t < 16; ++t) {
      if (t < 15)
        bb[(t + 1) & 1] =
            gather_rowmajor(x2h, DHEAD, colBase + (t + 1) * 16 + nc, k0, lane);
      acc[t] = __builtin_amdgcn_wmma_f32_16x16x32_bf16(
          false, a, false, bb[t & 1], (short)0, acc[t], false, false);
    }
  }

#pragma unroll
  for (int t = 0; t < 16; ++t) {
    const int col = colBase + t * 16 + nc;
#pragma unroll
    for (int r = 0; r < 8; ++r) {
      const int row = rb + r;
      sc[row][col] = (__bf16)(acc[t][r] * SCALE +
                              x3h[(size_t)(rowBase + row) * SEQ + col]);
    }
  }
  __syncthreads();

  // ---------------- Softmax over each row of 2048 ----------------
  {
    const int row = tid >> 4;                   // 16 rows x 16 segments
    const int seg = tid & 15;
    const int c0  = seg * 128;

    float m = -3.402823466e+38f;
    for (int c = 0; c < 128; ++c) m = fmaxf(m, (float)sc[row][c0 + c]);
    red[row][seg] = m;
    __syncthreads();

    float rm = red[row][0];
#pragma unroll
    for (int i = 1; i < 16; ++i) rm = fmaxf(rm, red[row][i]);
    __syncthreads();                            // all reads of red done

    float lsum = 0.f;
    for (int c = 0; c < 128; ++c) {
      float e = __expf((float)sc[row][c0 + c] - rm);
      sc[row][c0 + c] = (__bf16)e;
      lsum += e;
    }
    red[row][seg] = lsum;
    __syncthreads();

    float rs = 0.f;
#pragma unroll
    for (int i = 0; i < 16; ++i) rs += red[row][i];
    const float inv = 1.0f / rs;
    for (int c = 0; c < 128; ++c)
      sc[row][c0 + c] = (__bf16)((float)sc[row][c0 + c] * inv);
  }
  __syncthreads();

  // ---------------- Phase B: out = attn * x3 ----------------
  {
    v8f z = {};
#pragma unroll
    for (int t = 0; t < 16; ++t) acc[t] = z;
  }

  for (int k0 = 0; k0 < SEQ; k0 += 32) {
    if (k0 + 32 < SEQ)                          // pull next x3 K-slab toward L2/L0
      __builtin_prefetch(&x3h[(size_t)(k0 + 32) * SEQ + colBase + lane], 0, 0);

    // A fragment: attn rows straight from bf16 LDS (2x 16B ds loads per lane)
    v16bf a;
#pragma unroll
    for (int j = 0; j < 8; ++j) a[j]     = sc[nc][k0 + kb + j];
#pragma unroll
    for (int j = 0; j < 8; ++j) a[8 + j] = sc[nc][k0 + kb + 16 + j];

    v16bf bb[2];
    bb[0] = gather_colmajor(x3h, SEQ, colBase + nc, k0, lane);
#pragma unroll
    for (int t = 0; t < 16; ++t) {
      if (t < 15)                               // prefetch next tile's B into the
        bb[(t + 1) & 1] =                       // other buffer: its loads/cvts
            gather_colmajor(x3h, SEQ,           // co-execute with this WMMA
                            colBase + (t + 1) * 16 + nc, k0, lane);
      acc[t] = __builtin_amdgcn_wmma_f32_16x16x32_bf16(
          false, a, false, bb[t & 1], (short)0, acc[t], false, false);
    }
  }

  // ---------------- Store 16x256 f32 output block per wave ----------------
#pragma unroll
  for (int t = 0; t < 16; ++t) {
    const int col = colBase + t * 16 + nc;
#pragma unroll
    for (int r = 0; r < 8; ++r) {
      const int row = rb + r;
      oh[(size_t)(rowBase + row) * SEQ + col] = acc[t][r];
    }
  }
}

extern "C" void kernel_launch(void* const* d_in, const int* in_sizes, int n_in,
                              void* d_out, int out_size, void* d_ws, size_t ws_size,
                              hipStream_t stream) {
  const float* x1 = (const float*)d_in[0];
  const float* x2 = (const float*)d_in[1];
  const float* x3 = (const float*)d_in[2];
  float* out = (float*)d_out;

  dim3 grid(SEQ / 16, NHEAD);                   // 128 row-blocks x 16 heads
  fused_attn_bias_gemm<<<grid, 256, 0, stream>>>(x1, x2, x3, out);
}
